// custom_layer_arcface_72713796322206
// MI455X (gfx1250) — matrix operand order repaired and checked by reading, not compiled, not run on hardware
//
#include <hip/hip_runtime.h>

// ---------------------------------------------------------------------------
// ArcFace head for MI455X (gfx1250, wave32, WMMA)
//   B=1024, D=512, C=100000
//   out = softmax( ||y||_row * ( l2norm_row(y) @ l2norm_col(W) + margin@target ) )
//
// GEMM: block tile 128Mx128N (8 waves, 4Mx2N), wave tile 32Mx64N
//       (2 M-tiles x 4 N-tiles, 8x v_wmma_f32_16x16x32_bf16 per 32-K step).
//       wn (bf16, 102 MB) stays L2-resident; 8 row-block passes stream it.
// ---------------------------------------------------------------------------

#define B_   1024
#define D_   512
#define C_   100000
#define CP_  100096            // C padded to 782*128 (multiple of 128-col block)
#define NT_  782               // number of 128-wide column blocks

#define COS_M 0.98877107793604206f   // cos(0.15)
#define SIN_M 0.14943813247359922f   // sin(0.15)
#define CLIP_EPS_ 1.0e-5f
#define NEG_BIG (-3.0e38f)

typedef __bf16 bf16x8_t  __attribute__((ext_vector_type(8)));
typedef __bf16 bf16x16_t __attribute__((ext_vector_type(16)));
typedef float  f32x8_t   __attribute__((ext_vector_type(8)));

__device__ __forceinline__ bf16x16_t pack16(bf16x8_t lo, bf16x8_t hi) {
  bf16x16_t r;
#pragma unroll
  for (int i = 0; i < 8; ++i) { r[i] = lo[i]; r[i + 8] = hi[i]; }
  return r;
}

// ---------------------------------------------------------------------------
// 1) Row-normalize y_pred -> bf16 xn [B, D]; store per-row L2 norm.
// ---------------------------------------------------------------------------
__global__ __launch_bounds__(256) void arc_xnorm_kernel(
    const float* __restrict__ y, __bf16* __restrict__ xn, float* __restrict__ rown)
{
  const int row = blockIdx.x;
  const int tid = threadIdx.x;
  const float v0 = y[(size_t)row * D_ + tid];
  const float v1 = y[(size_t)row * D_ + 256 + tid];
  float sq = v0 * v0 + v1 * v1;
#pragma unroll
  for (int off = 1; off < 32; off <<= 1) sq += __shfl_xor(sq, off, 32);
  __shared__ float sm[8];
  if ((tid & 31) == 0) sm[tid >> 5] = sq;
  __syncthreads();
  float tot = 0.0f;
#pragma unroll
  for (int i = 0; i < 8; ++i) tot += sm[i];
  const float inv = rsqrtf(fmaxf(tot, 1.0e-12f));
  xn[(size_t)row * D_ + tid]       = (__bf16)(v0 * inv);
  xn[(size_t)row * D_ + 256 + tid] = (__bf16)(v1 * inv);
  if (tid == 0) rown[row] = sqrtf(tot);
}

// ---------------------------------------------------------------------------
// 2) Column-normalize W -> bf16 wn [D, CP_] (zero-padded columns >= C).
//    wn is 102 MB -> resident in the 192 MB L2 for the GEMM.
// ---------------------------------------------------------------------------
__global__ __launch_bounds__(256) void arc_wnorm_kernel(
    const float* __restrict__ W, __bf16* __restrict__ wn)
{
  const int c = blockIdx.x * 256 + threadIdx.x;
  if (c >= CP_) return;
  if (c < C_) {
    float sq = 0.0f;
    for (int d = 0; d < D_; ++d) {
      const float v = W[(size_t)d * C_ + c];
      sq += v * v;
    }
    const float inv = rsqrtf(fmaxf(sq, 1.0e-12f));
    for (int d = 0; d < D_; ++d)
      wn[(size_t)d * CP_ + c] = (__bf16)(W[(size_t)d * C_ + c] * inv);
  } else {
    for (int d = 0; d < D_; ++d)
      wn[(size_t)d * CP_ + c] = (__bf16)0.0f;
  }
}

// ---------------------------------------------------------------------------
// 3) WMMA GEMM + fused ArcFace margin + norm scale + online-softmax partials.
//    Block: 256 threads = 8 waves arranged 4(M) x 2(N).
//    Wave: 32Mx64N tile = 2 M-tiles x 4 N-tiles, 8 accumulators.
//    Per 32-K step: 4 A-loads (b128) + 64 scalar bf16 B-loads feed 8 WMMAs.
// ---------------------------------------------------------------------------
__global__ __launch_bounds__(256) void arc_gemm_kernel(
    const __bf16* __restrict__ xn, const __bf16* __restrict__ wn,
    const float* __restrict__ rown, const int* __restrict__ ytrue,
    float* __restrict__ out, float* __restrict__ partM, float* __restrict__ partS)
{
  const int wave = threadIdx.x >> 5;
  const int lane = threadIdx.x & 31;
  const int half = lane >> 4;     // 0: lanes 0-15, 1: lanes 16-31
  const int l15  = lane & 15;
  const int wm  = wave >> 1;      // 0..3  -> 32-row group within block
  const int wn2 = wave & 1;       // 0..1  -> 64-col group within block
  const int m0 = blockIdx.y * 128 + wm * 32;
  const int c0 = blockIdx.x * 128 + wn2 * 64;

  f32x8_t acc[2][4] = {};

  // A-fragment (16x32 bf16): lane row = l15; lanes<16 hold K {0..7,16..23},
  // lanes>=16 hold K {8..15,24..31}  -> two contiguous 16B chunks per lane.
  const __bf16* arow0 = xn + (size_t)(m0 + l15) * D_;
  const __bf16* arow1 = arow0 + (size_t)16 * D_;

  for (int k0 = 0; k0 < D_; k0 += 32) {
    const bf16x16_t a0 = pack16(*(const bf16x8_t*)(arow0 + k0 + half * 8),
                                *(const bf16x8_t*)(arow0 + k0 + 16 + half * 8));
    const bf16x16_t a1 = pack16(*(const bf16x8_t*)(arow1 + k0 + half * 8),
                                *(const bf16x8_t*)(arow1 + k0 + 16 + half * 8));

    // B-fragment (32x16 bf16): lane col N = l15; K {8h..8h+7, 16+8h..16+8h+7} (matches A).
    const __bf16* brow = wn + (size_t)(k0 + half * 8) * CP_ + c0 + l15;
#pragma unroll
    for (int t = 0; t < 4; ++t) {
      bf16x16_t b;
#pragma unroll
      for (int i = 0; i < 8; ++i) {
        b[i]     = brow[(size_t)i * CP_ + t * 16];
        b[i + 8] = brow[(size_t)(16 + i) * CP_ + t * 16];
      }
      acc[0][t] = __builtin_amdgcn_wmma_f32_16x16x32_bf16(
          false, a0, false, b, (short)0, acc[0][t], false, false);
      acc[1][t] = __builtin_amdgcn_wmma_f32_16x16x32_bf16(
          false, a1, false, b, (short)0, acc[1][t], false, false);
    }
  }

  // Epilogue. C/D layout: VGPR r -> M = r + 8*half; N = c0 + t*16 + l15.
#pragma unroll
  for (int s = 0; s < 2; ++s) {
#pragma unroll
    for (int r = 0; r < 8; ++r) {
      const int m = m0 + s * 16 + r + half * 8;
      const float nr = rown[m];
      const int tgt = ytrue[m];
      float vals[4];
      float rowmax = NEG_BIG;
#pragma unroll
      for (int t = 0; t < 4; ++t) {
        const int n = c0 + t * 16 + l15;
        float v;
        if (n < C_) {
          float cth = acc[s][t][r];
          if (n == tgt) {
            // cos(theta + m) = c*cos(m) - sqrt(1-c^2)*sin(m), with c clipped
            const float cc = fminf(fmaxf(cth, -1.0f + CLIP_EPS_), 1.0f - CLIP_EPS_);
            const float sn = sqrtf(fmaxf(1.0f - cc * cc, 0.0f));
            const float ncos = cc * COS_M - sn * SIN_M;
            cth += (ncos - cc);
          }
          v = cth * nr;
          out[(size_t)m * C_ + n] = v;
        } else {
          v = NEG_BIG;  // padded column: excluded from softmax
        }
        vals[t] = v;
        rowmax = fmaxf(rowmax, v);
      }
      // Reduce across the 16 lanes of this half-wave (xor 1,2,4,8 stays in half)
#pragma unroll
      for (int off = 1; off < 16; off <<= 1)
        rowmax = fmaxf(rowmax, __shfl_xor(rowmax, off, 32));
      float se = 0.0f;
#pragma unroll
      for (int t = 0; t < 4; ++t) se += __expf(vals[t] - rowmax);
#pragma unroll
      for (int off = 1; off < 16; off <<= 1)
        se += __shfl_xor(se, off, 32);
      if (l15 == 0) {
        partM[(size_t)m * NT_ + blockIdx.x] = rowmax;
        partS[(size_t)m * NT_ + blockIdx.x] = se;
      }
    }
  }
}

// ---------------------------------------------------------------------------
// 4) Combine per-block softmax partials -> per-row global max & sum.
// ---------------------------------------------------------------------------
__global__ __launch_bounds__(256) void arc_combine_kernel(
    const float* __restrict__ partM, const float* __restrict__ partS,
    float* __restrict__ rowM, float* __restrict__ rowS)
{
  const int row = blockIdx.x;
  const int tid = threadIdx.x;
  __shared__ float sm[8];

  float mx = NEG_BIG;
  for (int i = tid; i < NT_; i += 256)
    mx = fmaxf(mx, partM[(size_t)row * NT_ + i]);
#pragma unroll
  for (int off = 1; off < 32; off <<= 1) mx = fmaxf(mx, __shfl_xor(mx, off, 32));
  if ((tid & 31) == 0) sm[tid >> 5] = mx;
  __syncthreads();
  float gm = sm[0];
#pragma unroll
  for (int i = 1; i < 8; ++i) gm = fmaxf(gm, sm[i]);
  __syncthreads();

  float s = 0.0f;
  for (int i = tid; i < NT_; i += 256) {
    const float pm = partM[(size_t)row * NT_ + i];
    s += partS[(size_t)row * NT_ + i] * __expf(pm - gm);  // -> 0 for padded tiles
  }
#pragma unroll
  for (int off = 1; off < 32; off <<= 1) s += __shfl_xor(s, off, 32);
  if ((tid & 31) == 0) sm[tid >> 5] = s;
  __syncthreads();
  if (tid == 0) {
    float tot = 0.0f;
#pragma unroll
    for (int i = 0; i < 8; ++i) tot += sm[i];
    rowM[row] = gm;
    rowS[row] = tot;
  }
}

// ---------------------------------------------------------------------------
// 5) Normalize: out = exp(logit - rowM) / rowS  (float4-vectorized).
// ---------------------------------------------------------------------------
__global__ __launch_bounds__(256) void arc_softmax_kernel(
    float* __restrict__ out, const float* __restrict__ rowM,
    const float* __restrict__ rowS)
{
  const int row = blockIdx.y;
  const int c = (blockIdx.x * 256 + threadIdx.x) * 4;
  if (c >= C_) return;
  const float gm = rowM[row];
  const float sinv = 1.0f / rowS[row];
  float4* p = (float4*)(out + (size_t)row * C_ + c);
  float4 v = *p;
  v.x = __expf(v.x - gm) * sinv;
  v.y = __expf(v.y - gm) * sinv;
  v.z = __expf(v.z - gm) * sinv;
  v.w = __expf(v.w - gm) * sinv;
  *p = v;
}

// ---------------------------------------------------------------------------
extern "C" void kernel_launch(void* const* d_in, const int* in_sizes, int n_in,
                              void* d_out, int out_size, void* d_ws, size_t ws_size,
                              hipStream_t stream)
{
  (void)in_sizes; (void)n_in; (void)out_size; (void)ws_size;
  const float* y_pred = (const float*)d_in[0];
  const int*   y_true = (const int*)d_in[1];
  const float* W      = (const float*)d_in[2];
  float* out = (float*)d_out;

  char* ws = (char*)d_ws;
  size_t off = 0;
  __bf16* xn   = (__bf16*)(ws + off); off += (size_t)B_ * D_ * sizeof(__bf16);   // 1 MB
  float*  rown = (float*)(ws + off);  off += (size_t)B_ * sizeof(float);
  float*  rowM = (float*)(ws + off);  off += (size_t)B_ * sizeof(float);
  float*  rowS = (float*)(ws + off);  off += (size_t)B_ * sizeof(float);
  off = (off + 255) & ~(size_t)255;
  float* partM = (float*)(ws + off);  off += (size_t)B_ * NT_ * sizeof(float);   // 3.2 MB
  float* partS = (float*)(ws + off);  off += (size_t)B_ * NT_ * sizeof(float);   // 3.2 MB
  off = (off + 255) & ~(size_t)255;
  __bf16* wn   = (__bf16*)(ws + off); off += (size_t)D_ * CP_ * sizeof(__bf16);  // 102 MB (L2-resident)

  arc_xnorm_kernel<<<B_, 256, 0, stream>>>(y_pred, xn, rown);
  arc_wnorm_kernel<<<CP_ / 256, 256, 0, stream>>>(W, wn);
  arc_gemm_kernel<<<dim3(NT_, B_ / 128), 256, 0, stream>>>(xn, wn, rown, y_true,
                                                           out, partM, partS);
  arc_combine_kernel<<<B_, 256, 0, stream>>>(partM, partS, rowM, rowS);
  arc_softmax_kernel<<<dim3((C_ / 4 + 255) / 256, B_), 256, 0, stream>>>(out, rowM, rowS);
}
